// NTFALayer_39943195852910
// MI455X (gfx1250) — matrix operand order repaired and checked by reading, not compiled, not run on hardware
//
#include <hip/hip_runtime.h>
#include <hip/hip_bf16.h>

typedef __attribute__((ext_vector_type(16))) _Float16 v16h;
typedef __attribute__((ext_vector_type(8)))  float    v8f;

#define L_IN      16384
#define NP        65533          // interpolated points per batch
#define ROWSTRIDE 65536          // padded row stride (halfs) for aligned b128 loads
#define NB        64
#define WIN       256
#define STEP      64
#define NWIN      1020           // (65533-256)/64 + 1
#define NROWS     (NB * NWIN)    // 65280 total windows
#define NTILES    (NROWS / 16)   // 4080 row tiles (exact)
#define CUBIC_A   (-0.75f)

union H8 { uint4 u; _Float16 h[8]; };

__device__ __forceinline__ v16h combine8(uint4 lo, uint4 hi) {
  H8 a, b; a.u = lo; b.u = hi;
  v16h r;
#pragma unroll
  for (int i = 0; i < 8; ++i) { r[i] = a.h[i]; r[i + 8] = b.h[i]; }
  return r;
}

// ---------------- DFT twiddle tables (symmetric 256x256, f16) ----------------
__global__ void tables_kernel(_Float16* __restrict__ cosT, _Float16* __restrict__ sinT) {
  int i = blockIdx.x * 256 + threadIdx.x;     // i = k*256 + t
  int k = i >> 8, t = i & 255;
  int prod = (k * t) & 255;                   // exact periodicity mod 256
  float th = (float)prod * (6.28318530717958647692f / 256.0f);
  float s, c;
  __sincosf(th, &s, &c);
  cosT[i] = (_Float16)c;
  sinT[i] = (_Float16)s;                      // sign of Im is irrelevant for |FFT|
}

// ---------------- cubic upsample + apply_kernels, f16 output -----------------
__device__ __forceinline__ float cubic_w(float t) {
  float at = fabsf(t);
  float w_near = ((CUBIC_A + 2.0f) * at - (CUBIC_A + 3.0f)) * at * at + 1.0f;
  float w_far  = ((CUBIC_A * at - 5.0f * CUBIC_A) * at + 8.0f * CUBIC_A) * at - 4.0f * CUBIC_A;
  return at <= 1.0f ? w_near : (at < 2.0f ? w_far : 0.0f);
}

__global__ void interp_kernel(const float* __restrict__ sig, _Float16* __restrict__ kw) {
  int p = blockIdx.x * 256 + threadIdx.x;     // 0 .. 65535
  int b = blockIdx.y;
  _Float16* row = kw + (size_t)b * ROWSTRIDE;
  if (p >= NP) { row[p] = (_Float16)0.0f; return; }   // zero the pad region
  int   q    = p >> 2;                        // i0 = floor(p * 0.25)
  float frac = (float)(p & 3) * 0.25f;
  const float* s = sig + (size_t)b * L_IN;
  float acc = 0.0f;
#pragma unroll
  for (int o = -1; o <= 2; ++o) {
    int idx = q + o;
    idx = idx < 0 ? 0 : (idx > (L_IN - 1) ? (L_IN - 1) : idx);
    acc += cubic_w(frac - (float)o) * s[idx];
  }
  float xp = acc + 1.3f;
  float g  = acc - 0.7f;
  float y  = 0.5f * xp * xp + 0.5f * expf(-0.5f * g * g);
  row[p] = (_Float16)y;
}

// -------- DFT-as-WMMA + |.| + row max + sigmoid mask + soft-min, fused -------
__global__ void __launch_bounds__(512)
dft_wmma_kernel(const _Float16* __restrict__ kw,
                const _Float16* __restrict__ cosT,
                const _Float16* __restrict__ sinT,
                float* __restrict__ out) {
  __shared__ _Float16 Asm[16][WIN];           // 16 windows x 256 samples (8 KB)
  __shared__ unsigned rowmaxU[16];

  const int tid  = threadIdx.x;               // 0..511, 16 wave32 waves
  const int tile = blockIdx.x;                // 0..4079
  const int row0 = tile * 16;

  // Cooperative stage of the A tile into LDS (512 lanes x 16B = 8 KB)
  {
    int rr = tid >> 5, part = tid & 31;
    int r = row0 + rr;
    int b = r / NWIN;
    int w = r - b * NWIN;
    const uint4* src = (const uint4*)(kw + (size_t)b * ROWSTRIDE + w * STEP + part * 8);
    *(uint4*)&Asm[rr][part * 8] = *src;
  }
  if (tid < 16) rowmaxU[tid] = 0u;
  __syncthreads();

  const int wave = tid >> 5;                  // freq tile 0..15
  const int lane = tid & 31;
  const int n    = lane & 15;
  const int hi   = lane >> 4;
  const int m    = n;                         // A-fragment row for this lane
  const int f0   = wave * 16;

  v8f re = {}; v8f im = {};
#pragma unroll
  for (int kb = 0; kb < 8; ++kb) {
    // A fragment: 16-bit A 16x32 layout (lane=M; K halves split by hi)
    int ka = kb * 32 + hi * 8;
    uint4 a0 = *(const uint4*)&Asm[m][ka];
    uint4 a1 = *(const uint4*)&Asm[m][ka + 16];
    v16h av = combine8(a0, a1);
    // B fragments: lane=N (freq f0+n, symmetric table row), elements = K split by hi like A
    const _Float16* cp = cosT + (size_t)(f0 + n) * 256 + ka;
    const _Float16* sp = sinT + (size_t)(f0 + n) * 256 + ka;
    v16h bc = combine8(*(const uint4*)cp, *(const uint4*)(cp + 16));
    v16h bs = combine8(*(const uint4*)sp, *(const uint4*)(sp + 16));
    re = __builtin_amdgcn_wmma_f32_16x16x32_f16(false, av, false, bc, (short)0, re, false, false);
    im = __builtin_amdgcn_wmma_f32_16x16x32_f16(false, av, false, bs, (short)0, im, false, false);
  }

  // Magnitudes: lane holds freq (f0+n) for 8 rows (rl = v + 8*hi)
  float Mv[8];
#pragma unroll
  for (int v = 0; v < 8; ++v)
    Mv[v] = sqrtf(re[v] * re[v] + im[v] * im[v]);

  // Max over the 16 freqs of this wave (xor masks < 16 stay within each half)
#pragma unroll
  for (int v = 0; v < 8; ++v) {
    float x = Mv[v];
    x = fmaxf(x, __shfl_xor(x, 1, 32));
    x = fmaxf(x, __shfl_xor(x, 2, 32));
    x = fmaxf(x, __shfl_xor(x, 4, 32));
    x = fmaxf(x, __shfl_xor(x, 8, 32));
    if (n == 0)  // lanes 0 and 16 carry the group max; M>=0 so uint order == float order
      atomicMax(&rowmaxU[v + 8 * hi], __float_as_uint(x));
  }
  __syncthreads();

  const int fcol = f0 + n;
#pragma unroll
  for (int v = 0; v < 8; ++v) {
    int   rl  = v + 8 * hi;
    float mx  = __uint_as_float(rowmaxU[rl]);
    float thr = 0.12f * mx;
    float mv  = Mv[v];
    float sg  = 1.0f / (1.0f + expf(thr - mv));       // sigmoid((M - thr) * 1)
    float mt  = mv * sg;
    // soft_minimum(mt, 0.9*mx, 1) = -logsumexp(-mt, -0.9*mx)
    float x1 = -mt, x2 = -0.9f * mx;
    float mm = fmaxf(x1, x2);
    float res = -(mm + logf(expf(x1 - mm) + expf(x2 - mm)));
    out[(size_t)(row0 + rl) * 256 + fcol] = res;
  }
}

extern "C" void kernel_launch(void* const* d_in, const int* in_sizes, int n_in,
                              void* d_out, int out_size, void* d_ws, size_t ws_size,
                              hipStream_t stream) {
  const float* signal = (const float*)d_in[0];
  _Float16* kwbuf = (_Float16*)d_ws;                                   // 8 MB
  _Float16* cosT  = (_Float16*)((char*)d_ws + (size_t)NB * ROWSTRIDE * sizeof(_Float16));
  _Float16* sinT  = cosT + 256 * 256;                                  // +128 KB each
  float* out = (float*)d_out;

  tables_kernel<<<256, 256, 0, stream>>>(cosT, sinT);
  interp_kernel<<<dim3(ROWSTRIDE / 256, NB), 256, 0, stream>>>(signal, kwbuf);
  dft_wmma_kernel<<<NTILES, 512, 0, stream>>>(kwbuf, cosT, sinT, out);
}
